// CFGNodeEncoderExpressionUpdateLayer_64665027608676
// MI455X (gfx1250) — matrix operand order repaired and checked by reading, not compiled, not run on hardware
//
#include <hip/hip_runtime.h>

// ---------------------------------------------------------------------------
// CFG node expression update (MI455X / gfx1250, wave32):
//   gate = sigmoid([prev|upd] @ W_forget + b_f)
//   out[idx] = gate*prev + (1-gate)*(upd @ W_proj + b_p), out = nodes elsewhere
// Single fused bf16 GEMM [M,768] x [768,512] -> [gate_logits | proj],
// f32 accumulation via v_wmma_f32_16x16x32_bf16. Weights pre-swizzled into
// WMMA B-fragment order; B staging is a contiguous ASYNC copy
// (global_load_async_to_lds_b128 / ASYNCcnt); zero-block (k<256 x proj cols)
// is skipped entirely; B fragments are software-pipelined.
// ---------------------------------------------------------------------------

typedef __attribute__((ext_vector_type(16))) __bf16 v16bf;
typedef __attribute__((ext_vector_type(8)))  float  v8f;
typedef int v4i_b128 __attribute__((vector_size(16)));   // async builtin payload
typedef __attribute__((address_space(1))) v4i_b128* gptr_b128;
typedef __attribute__((address_space(3))) v4i_b128* lptr_b128;

#define D_NODE 256
#define D_EXPR 512
#define K_TOT  768   // D_NODE + D_EXPR
#define N_OUT  512   // 256 gate cols + 256 proj cols
#define M_TILE 128
#define KC     32    // one WMMA K step per chunk
#define BLOCK  512   // 16 waves of 32
#define CHUNK_ELEMS (KC * N_OUT)   // 16384 bf16 = 32KB per K-chunk

__device__ __forceinline__ __bf16 f32_to_bf16_rne(float f) {
  unsigned u = __builtin_bit_cast(unsigned, f);
  unsigned r = (u + 0x7FFFu + ((u >> 16) & 1u)) >> 16;
  unsigned short s = (unsigned short)r;
  return __builtin_bit_cast(__bf16, s);
}

// pack two f32 -> two bf16 (one dword); prefer HW packed convert if present
__device__ __forceinline__ unsigned pack2_bf16(float a, float b) {
#if __has_builtin(__builtin_amdgcn_cvt_pk_bf16_f32)
  auto r = __builtin_amdgcn_cvt_pk_bf16_f32(a, b);
  unsigned u;
  __builtin_memcpy(&u, &r, 4);
  return u;
#else
  unsigned lo = __builtin_bit_cast(unsigned short, f32_to_bf16_rne(a));
  unsigned hi = __builtin_bit_cast(unsigned short, f32_to_bf16_rne(b));
  return lo | (hi << 16);
#endif
}

// ---------------- out <- nodes (float4 copy) ----------------
__global__ void k_copy(const float* __restrict__ nodes, float* __restrict__ out,
                       long total4) {
  long i = (long)blockIdx.x * blockDim.x + threadIdx.x;
  if (i < total4) {
    reinterpret_cast<float4*>(out)[i] =
        reinterpret_cast<const float4*>(nodes)[i];
  }
}

// ---- build fused bf16 weight matrix, PRE-SWIZZLED into B-fragment order ----
// Logical W(k, n): n<256 -> W_forget[k][n]; n>=256 -> (k<256 ? 0 : W_proj).
// Storage: Wc[ ((kc*32 + nsub)*32 + lane)*16 + j ] where per 16x16x32 bf16 B:
//   lane = ((kr>=16)?16:0) + (n%16),  j = kr%16,  kr = k%32, kc = k/32,
//   nsub = n/16.  => each lane's fragment is 16 contiguous bf16 (32B).
__global__ void k_weights(const float* __restrict__ Wf, const float* __restrict__ Wp,
                          __bf16* __restrict__ Wc) {
  int gid = blockIdx.x * blockDim.x + threadIdx.x;   // 768*512 threads
  if (gid >= K_TOT * N_OUT) return;
  int j    = gid & 15;
  int l    = (gid >> 4) & 31;
  int nsub = (gid >> 9) & 31;
  int kc   = gid >> 14;
  int k = kc * KC + ((l >> 4) << 4) + j;
  int n = nsub * 16 + (l & 15);
  float v;
  if (n < D_NODE) {
    v = Wf[k * D_NODE + n];
  } else if (k >= D_NODE) {
    v = Wp[(k - D_NODE) * D_NODE + (n - D_NODE)];
  } else {
    v = 0.0f;
  }
  Wc[gid] = f32_to_bf16_rne(v);
}

// ---------------- ordered mask compaction (single block, 1024 thr) -----------
__global__ void k_index(const unsigned char* __restrict__ mask, int N,
                        int* __restrict__ idx, int M) {
  __shared__ int woff[32];
  __shared__ int s_tot;
  __shared__ int s_base;
  if (threadIdx.x == 0) s_base = 0;
  __syncthreads();
  const int lane = threadIdx.x & 31;
  const int wid  = threadIdx.x >> 5;
  for (int b = 0; b < N; b += 1024) {
    int i = b + (int)threadIdx.x;
    int m = (i < N) ? (mask[i] != 0) : 0;
    unsigned long long bal = __ballot(m);
    int pre  = __popcll(bal & ((1ull << lane) - 1ull));
    int wcnt = __popcll(bal & 0xFFFFFFFFull);
    if (lane == 0) woff[wid] = wcnt;
    __syncthreads();
    if (threadIdx.x == 0) {
      int acc = 0;
      for (int w = 0; w < 32; ++w) { int c = woff[w]; woff[w] = acc; acc += c; }
      s_tot = acc;
    }
    __syncthreads();
    int pos = s_base + woff[wid] + pre;
    if (m && pos < M) idx[pos] = i;
    __syncthreads();
    if (threadIdx.x == 0) s_base += s_tot;
    __syncthreads();
  }
}

// ---------------- main fused WMMA GEMM + gated scatter -----------------------
// Block: 128 M-rows, all 512 out cols. 16 waves; wave (rgrp=w>>1, cgrp=w&1)
// owns rows [rgrp*16,+16) and cols [cgrp*128,+128) of BOTH halves.
__global__ void __launch_bounds__(BLOCK)
k_main(const float* __restrict__ nodes, const float* __restrict__ upd,
       const int* __restrict__ idxbuf, const __bf16* __restrict__ Wc,
       const float* __restrict__ bF, const float* __restrict__ bP,
       float* __restrict__ out, int N, int M) {
  __shared__ int s_idx[M_TILE];
  __shared__ alignas(32) __bf16 sA[(M_TILE / 16) * 512];  // 8 A-fragments, 8KB
  __shared__ alignas(32) __bf16 sB[CHUNK_ELEMS];          // 1 B K-chunk, 32KB

  const int tid  = threadIdx.x;
  const int lane = tid & 31;
  const int wave = tid >> 5;
  const int rgrp = wave >> 1;              // 0..7
  const int cgrp = wave & 1;               // 0..1
  const int m0   = blockIdx.x * M_TILE;

  if (tid < M_TILE) {
    int r = m0 + tid;
    int nr = (r < M) ? idxbuf[r] : 0;
    s_idx[tid] = ((unsigned)nr < (unsigned)N) ? nr : 0;
  }
  __syncthreads();

  v8f accG[8], accP[8];
#pragma unroll
  for (int t = 0; t < 8; ++t) { accG[t] = (v8f)0.0f; accP[t] = (v8f)0.0f; }

  const int arow = tid >> 2;               // 0..127 (BLOCK=512)
  const int akg  = (tid & 3) * 8;          // 0,8,16,24
  // A-fragment swizzle target for this thread's 8 k-values (all same lane/run):
  const int adst = (arow >> 4) * 512 + ((arow & 15) + ((akg & 16) ? 16 : 0)) * 16 +
                   ((akg & 8) ? 8 : 0);
  const __bf16* sBg = &sB[(cgrp * 8) * 512 + lane * 16];        // gate frag base
  const __bf16* sBp = &sB[(16 + cgrp * 8) * 512 + lane * 16];   // proj frag base

  for (int kc = 0; kc < K_TOT; kc += KC) {
    const bool full = (kc >= D_NODE);      // proj rows of Wc are nonzero
    // ---- stage A chunk (gather + f32->bf16, written in fragment order) ----
    {
      const float* src;
      if (!full) {
        src = nodes + (size_t)s_idx[arow] * D_NODE + (kc + akg);
      } else {
        size_t ur = (size_t)((m0 + arow < M) ? (m0 + arow) : 0);
        src = upd + ur * D_EXPR + (kc - D_NODE + akg);
      }
      uint4 pk;
      pk.x = pack2_bf16(src[0], src[1]);
      pk.y = pack2_bf16(src[2], src[3]);
      pk.z = pack2_bf16(src[4], src[5]);
      pk.w = pack2_bf16(src[6], src[7]);
      *reinterpret_cast<uint4*>(&sA[adst]) = pk;
    }
    // ---- stage B chunk: contiguous async copy of pre-swizzled weights ----
    // For kc<256 the proj half of the chunk is all-zero: copy/compute only gate.
    {
      const __bf16* gB = Wc + ((size_t)(kc >> 5) << 14);  // chunk base
      const int nIter = full ? 4 : 2;                     // b128s per thread
#if __has_builtin(__builtin_amdgcn_global_load_async_to_lds_b128)
      for (int i = 0; i < nIter; ++i) {
        int q = (i * BLOCK + tid) * 8;
        __builtin_amdgcn_global_load_async_to_lds_b128(
            (gptr_b128)(gB + q), (lptr_b128)(&sB[q]), 0, 0);
      }
#if __has_builtin(__builtin_amdgcn_s_wait_asynccnt)
      __builtin_amdgcn_s_wait_asynccnt(0);
#else
      asm volatile("s_wait_asynccnt 0x0" ::: "memory");
#endif
#else
      for (int i = 0; i < nIter; ++i) {
        int q = (i * BLOCK + tid) * 8;
        *reinterpret_cast<uint4*>(&sB[q]) =
            *reinterpret_cast<const uint4*>(gB + q);
      }
#endif
    }
    if (kc + KC < K_TOT)  // pull next weight chunk toward the WGP (L2 hit)
      __builtin_prefetch(Wc + ((size_t)((kc >> 5) + 1) << 14), 0, 0);
    __syncthreads();

    // ---- fragments: one contiguous 32B LDS read per lane, pipelined ----
    v16bf afrag = *reinterpret_cast<const v16bf*>(&sA[rgrp * 512 + lane * 16]);
    if (full) {
      v16bf bG = *reinterpret_cast<const v16bf*>(sBg);
      v16bf bP2 = *reinterpret_cast<const v16bf*>(sBp);
#pragma unroll
      for (int t = 0; t < 8; ++t) {
        v16bf bGn, bPn;
        if (t < 7) {
          bGn = *reinterpret_cast<const v16bf*>(sBg + (t + 1) * 512);
          bPn = *reinterpret_cast<const v16bf*>(sBp + (t + 1) * 512);
        }
        accG[t] = __builtin_amdgcn_wmma_f32_16x16x32_bf16(
            false, afrag, false, bG, (short)0, accG[t], false, false);
        accP[t] = __builtin_amdgcn_wmma_f32_16x16x32_bf16(
            false, afrag, false, bP2, (short)0, accP[t], false, false);
        if (t < 7) { bG = bGn; bP2 = bPn; }
      }
    } else {
      v16bf bG = *reinterpret_cast<const v16bf*>(sBg);
#pragma unroll
      for (int t = 0; t < 8; ++t) {
        v16bf bGn;
        if (t < 7) bGn = *reinterpret_cast<const v16bf*>(sBg + (t + 1) * 512);
        accG[t] = __builtin_amdgcn_wmma_f32_16x16x32_bf16(
            false, afrag, false, bG, (short)0, accG[t], false, false);
        if (t < 7) bG = bGn;
      }
    }
    __syncthreads();
  }

  // ---- epilogue: gate = sigmoid(G+bF); out = gate*prev + (1-gate)*(P+bP) ----
  const int nlo = lane & 15;
  const int mhi = (lane >> 4) * 8;     // C/D layout: lanes 16-31 hold M=8..15
#pragma unroll
  for (int t = 0; t < 8; ++t) {
    int n = cgrp * 128 + t * 16 + nlo;  // 0..255 within each half
    float bf = bF[n];
    float bp = bP[n];
#pragma unroll
    for (int v = 0; v < 8; ++v) {
      int mloc = mhi + v;
      int row  = m0 + rgrp * 16 + mloc;
      if (row < M) {
        int nr = s_idx[rgrp * 16 + mloc];
        size_t off = (size_t)nr * D_NODE + n;
        float prev = nodes[off];
        float g = 1.0f / (1.0f + __expf(-(accG[t][v] + bf)));
        float pj = accP[t][v] + bp;
        out[off] = g * prev + (1.0f - g) * pj;
      }
    }
  }
}

// ---------------------------------------------------------------------------
extern "C" void kernel_launch(void* const* d_in, const int* in_sizes, int n_in,
                              void* d_out, int out_size, void* d_ws, size_t ws_size,
                              hipStream_t stream) {
  const float*         nodes = (const float*)d_in[0];
  const float*         upd   = (const float*)d_in[1];
  const unsigned char* mask  = (const unsigned char*)d_in[2];  // jnp.bool_: 1 byte
  const float*         Wf    = (const float*)d_in[3];
  const float*         bF    = (const float*)d_in[4];
  const float*         Wp    = (const float*)d_in[5];
  const float*         bP    = (const float*)d_in[6];
  float*               out   = (float*)d_out;

  const int N = in_sizes[0] / D_NODE;   // 400000
  const int M = in_sizes[1] / D_EXPR;   // 200000

  // workspace layout: [ Wc bf16 768*512 (swizzled) | idx int32 M ]
  __bf16* Wc  = (__bf16*)d_ws;
  int*    idx = (int*)((char*)d_ws + (size_t)K_TOT * N_OUT * sizeof(__bf16));

  long total4 = (long)N * D_NODE / 4;
  k_copy<<<(unsigned)((total4 + 255) / 256), 256, 0, stream>>>(nodes, out, total4);
  k_weights<<<(K_TOT * N_OUT + 255) / 256, 256, 0, stream>>>(Wf, Wp, Wc);
  k_index<<<1, 1024, 0, stream>>>(mask, N, idx, M);
  k_main<<<(M + M_TILE - 1) / M_TILE, BLOCK, 0, stream>>>(
      nodes, upd, idx, Wc, bF, bP, out, N, M);
}
